// MoE_2104533975402
// MI455X (gfx1250) — matrix operand order repaired and checked by reading, not compiled, not run on hardware
//
#include <hip/hip_runtime.h>
#include <hip/hip_bf16.h>
#include <stdint.h>

// ---------------- problem constants ----------------
#define DD 1024          // model dim
#define EE 8             // experts
#define HH 2730          // ffn hidden (int(8*1024/3))
#define HP 2752          // padded hidden, 64*43, zero-padded
#define NN 8192          // tokens (B*T = 4*2048)
#define BM 128           // M tile (slots)
#define BN 64            // N tile
#define BK 32            // K tile (= WMMA K for bf16)
#define NUM_MTILES 136   // max sum of ceil(cnt_e/BM) over experts
#define CAP (NUM_MTILES * BM)   // 17408 padded slots

#if defined(__has_builtin)
#if __has_builtin(__builtin_amdgcn_tensor_load_to_lds)
#define HAVE_TDM 1
#endif
#endif
#ifndef HAVE_TDM
#define HAVE_TDM 0
#endif

typedef __bf16 bf16_t;
typedef __attribute__((ext_vector_type(8)))  __bf16 bf16x8;
typedef __attribute__((ext_vector_type(16))) __bf16 bf16x16;
typedef __attribute__((ext_vector_type(8)))  float  f32x8;
typedef __attribute__((ext_vector_type(4)))  unsigned int u32x4;
typedef __attribute__((ext_vector_type(4)))  int i32x4;
typedef __attribute__((ext_vector_type(8)))  int i32x8;

__device__ __forceinline__ bf16x16 cat8(bf16x8 a, bf16x8 b) {
  return __builtin_shufflevector(a, b, 0,1,2,3,4,5,6,7,8,9,10,11,12,13,14,15);
}

// ---- CDNA5 async global->LDS copy (ASYNCcnt-tracked, no VGPR round trip) ----
__device__ __forceinline__ void async_b128(unsigned lds_byte_off, const void* gaddr) {
  asm volatile("global_load_async_to_lds_b128 %0, %1, off"
               :: "v"(lds_byte_off), "v"(gaddr)
               : "memory");
}

template <int N>
__device__ __forceinline__ void wait_async() {
#if __has_builtin(__builtin_amdgcn_s_wait_asynccnt)
  __builtin_amdgcn_s_wait_asynccnt(N);
#else
  asm volatile("s_wait_asynccnt %0" :: "n"(N) : "memory");
#endif
}

template <int N>
__device__ __forceinline__ void wait_tensor() {
#if __has_builtin(__builtin_amdgcn_s_wait_tensorcnt)
  __builtin_amdgcn_s_wait_tensorcnt((short)N);
#else
  asm volatile("s_wait_tensorcnt %0" :: "n"(N) : "memory");
#endif
}

__device__ __forceinline__ unsigned lds_off32(const void* p) {
  return (unsigned)(uintptr_t)p;   // LDS aperture: addr[31:0] is the LDS byte address
}

#if HAVE_TDM
// ---- Tensor Data Mover: 2D tile (rows x cols bf16, row stride in elements) ----
// D# per CDNA5 ISA ch.8: group0 {count, lds_addr, global_addr, type=2},
// group1 {data_size=2B, tensor_dim==tile_dim (no OOB), tile dims, dim0 stride}.
__device__ __forceinline__ void tdm_load_2d(unsigned lds_addr, const void* gaddr,
                                            unsigned rows, unsigned cols_elems,
                                            unsigned stride_elems) {
  const unsigned long long ga = (unsigned long long)(uintptr_t)gaddr;
  const unsigned td0 = cols_elems;   // tensor_dim0 == tile_dim0 -> never OOB
  const unsigned td1 = rows;         // tensor_dim1 == tile_dim1
  u32x4 g0 = { 1u,                                        // count=1 valid descriptor
               lds_addr,                                  // lds_addr [63:32]
               (unsigned)ga,                              // global_addr [95:64]
               ((unsigned)(ga >> 32) & 0x01FFFFFFu) | (2u << 30) };  // addr[56:32] | type=2
  i32x8 g1 = { (int)0x00010000u,                          // data_size=1 (2 bytes)
               (int)((td0 & 0xFFFFu) << 16),              // tensor_dim0[15:0] @ [63:48]
               (int)((td0 >> 16) | ((td1 & 0xFFFFu) << 16)),
               (int)((td1 >> 16) | (cols_elems << 16)),   // tile_dim0 @ [127:112]
               (int)rows,                                 // tile_dim1 (tile_dim2=0)
               (int)stride_elems,                         // tensor_dim0_stride[31:0]
               0, 0 };
  i32x4 gz = { 0, 0, 0, 0 };
#if __clang_major__ >= 23
  i32x8 gz8 = { 0, 0, 0, 0, 0, 0, 0, 0 };
  __builtin_amdgcn_tensor_load_to_lds(g0, g1, gz, gz, gz8, 0);
#else
  __builtin_amdgcn_tensor_load_to_lds(g0, g1, gz, gz, 0);
#endif
}
#endif  // HAVE_TDM

// ---------------- conversion kernels ----------------
__global__ __launch_bounds__(256) void k_convert_x(const float* __restrict__ x,
                                                   bf16_t* __restrict__ xb) {
  const int i = (blockIdx.x * 256 + threadIdx.x) * 4;
  float4 v = *(const float4*)(x + i);
  xb[i + 0] = (bf16_t)v.x;
  xb[i + 1] = (bf16_t)v.y;
  xb[i + 2] = (bf16_t)v.z;
  xb[i + 3] = (bf16_t)v.w;
}

// src: [E][D][H] fp32  ->  dst: [E][HP][D] bf16 (transposed, zero pad n>=H)
__global__ __launch_bounds__(256) void k_convert_w13t(const float* __restrict__ src,
                                                      bf16_t* __restrict__ dst) {
  const long idx = (long)blockIdx.x * 256 + threadIdx.x;   // < E*D*HP
  const int n = (int)(idx % HP);
  const int k = (int)((idx / HP) % DD);
  const int e = (int)(idx / ((long)HP * DD));
  const float v = (n < HH) ? src[((size_t)e * DD + k) * HH + n] : 0.0f;
  dst[((size_t)e * HP + n) * DD + k] = (bf16_t)v;
}

// src: [E][H][D] fp32  ->  dst: [E][D][HP] bf16 (transposed, zero pad hk>=H)
__global__ __launch_bounds__(256) void k_convert_w2t(const float* __restrict__ src,
                                                     bf16_t* __restrict__ dst) {
  const long idx = (long)blockIdx.x * 256 + threadIdx.x;   // < E*D*HP
  const int hk = (int)(idx % HP);
  const int d  = (int)((idx / HP) % DD);
  const int e  = (int)(idx / ((long)HP * DD));
  const float v = (hk < HH) ? src[((size_t)e * HH + hk) * DD + d] : 0.0f;
  dst[((size_t)e * DD + d) * HP + hk] = (bf16_t)v;
}

// ---------------- router: one wave32 per token ----------------
__global__ __launch_bounds__(256) void k_router(const float* __restrict__ x,
                                                const float* __restrict__ Wr,
                                                int* __restrict__ topi,
                                                float* __restrict__ topv) {
  const int lane = threadIdx.x & 31;
  const int wave = threadIdx.x >> 5;
  const int n = blockIdx.x * 8 + wave;
  const float* __restrict__ xr = x + (size_t)n * DD;

  float acc[EE] = {0.f, 0.f, 0.f, 0.f, 0.f, 0.f, 0.f, 0.f};
  for (int t = 0; t < DD / 32; ++t) {
    const int d = t * 32 + lane;
    const float xv = xr[d];
    const float4* wp = (const float4*)(Wr + (size_t)d * EE);
    const float4 w0 = wp[0], w1 = wp[1];
    acc[0] += xv * w0.x; acc[1] += xv * w0.y;
    acc[2] += xv * w0.z; acc[3] += xv * w0.w;
    acc[4] += xv * w1.x; acc[5] += xv * w1.y;
    acc[6] += xv * w1.z; acc[7] += xv * w1.w;
  }
#pragma unroll
  for (int m = 16; m >= 1; m >>= 1) {
#pragma unroll
    for (int e = 0; e < EE; ++e) acc[e] += __shfl_xor(acc[e], m, 32);
  }
  float mx = acc[0];
#pragma unroll
  for (int e = 1; e < EE; ++e) mx = fmaxf(mx, acc[e]);
  float p[EE], s = 0.f;
#pragma unroll
  for (int e = 0; e < EE; ++e) { p[e] = __expf(acc[e] - mx); s += p[e]; }
  const float inv = 1.f / s;
  int i1 = 0; float v1 = p[0];
#pragma unroll
  for (int e = 1; e < EE; ++e) if (p[e] > v1) { v1 = p[e]; i1 = e; }
  int i2 = (i1 == 0) ? 1 : 0; float v2 = p[i2];
#pragma unroll
  for (int e = 0; e < EE; ++e)
    if (e != i1 && p[e] > v2) { v2 = p[e]; i2 = e; }
  if (lane == 0) {
    topi[2 * n + 0] = i1; topi[2 * n + 1] = i2;
    topv[2 * n + 0] = v1 * inv; topv[2 * n + 1] = v2 * inv;
  }
}

// ---------------- deterministic slot assignment (single block) ----------------
__global__ void k_assign(const int* __restrict__ topi,
                         int* __restrict__ slot_token,
                         int* __restrict__ slot_idx,
                         int* __restrict__ tile_expert) {
  __shared__ int cnt[EE], offs[EE];
  const int t = threadIdx.x;   // 32 threads
  for (int s = t; s < CAP; s += 32) slot_token[s] = 0;   // pads read token 0
  __syncthreads();
  if (t < EE) {
    int c = 0;
    for (int n = 0; n < NN; ++n)
      c += (topi[2 * n] == t) + (topi[2 * n + 1] == t);
    cnt[t] = c;
  }
  __syncthreads();
  if (t == 0) {
    int off = 0, tt = 0;
    for (int e = 0; e < EE; ++e) {
      offs[e] = off;
      const int ntile = (cnt[e] + BM - 1) / BM;
      off += ntile * BM;
      for (int i = 0; i < ntile; ++i) tile_expert[tt++] = e;
    }
    while (tt < NUM_MTILES) tile_expert[tt++] = 0;
  }
  __syncthreads();
  if (t < EE) {
    int s = offs[t];
    for (int n = 0; n < NN; ++n) {
#pragma unroll
      for (int k = 0; k < 2; ++k) {
        if (topi[2 * n + k] == t) {
          slot_token[s] = n;
          slot_idx[2 * n + k] = s;
          ++s;
        }
      }
    }
  }
}

// ---------------- pass 1: h = silu(X*W1) * (X*W3)  (grouped by expert) ----------------
__global__ __launch_bounds__(256) void k_ffn1(const bf16_t* __restrict__ Xb,
                                              const bf16_t* __restrict__ W1bT,
                                              const bf16_t* __restrict__ W3bT,
                                              const int* __restrict__ slot_token,
                                              const int* __restrict__ tile_expert,
                                              bf16_t* __restrict__ hbuf) {
  __shared__ alignas(16) bf16_t As[2][BM * BK];
  __shared__ alignas(16) bf16_t B1s[2][BN * BK];
  __shared__ alignas(16) bf16_t B3s[2][BN * BK];

  const int tid = threadIdx.x;
  const int lane = tid & 31;
  const int wave = tid >> 5;
  const int wm = (wave & 3) * 32;   // wave row offset (4 waves down)
  const int wn = (wave >> 2) * 32;  // wave col offset (2 waves across)
  const int hi = lane >> 4;
  const int lrow = lane & 15;

  const int slotBase = blockIdx.x * BM;
  const int colBase = blockIdx.y * BN;
  const int e = tile_expert[blockIdx.x];

  // A staging: 128 rows x 32 K, token-gathered, 2 lanes/row, 32B each (async per-lane)
  const int arow = tid >> 1, ahalf = tid & 1;
  const int token = slot_token[slotBase + arow];
  const uint4* __restrict__ asrc = (const uint4*)(Xb + (size_t)token * DD) + ahalf * 2;

  unsigned ldsA[2];
#pragma unroll
  for (int b = 0; b < 2; ++b)
    ldsA[b] = lds_off32(&As[b][arow * BK + ahalf * 16]);

#if HAVE_TDM
  // B staging: whole 64x32 bf16 tile per buffer via one TDM descriptor (wave 0)
  const bool tdmw = (wave == 0);
  const bf16_t* b1base = W1bT + ((size_t)e * HP + colBase) * DD;
  const bf16_t* b3base = W3bT + ((size_t)e * HP + colBase) * DD;
  const unsigned ldsB1buf[2] = { lds_off32(&B1s[0][0]), lds_off32(&B1s[1][0]) };
  const unsigned ldsB3buf[2] = { lds_off32(&B3s[0][0]), lds_off32(&B3s[1][0]) };
#else
  const int brow = tid >> 2, bq = tid & 3;
  const uint4* __restrict__ b1src =
      (const uint4*)(W1bT + ((size_t)e * HP + colBase + brow) * DD) + bq;
  const uint4* __restrict__ b3src =
      (const uint4*)(W3bT + ((size_t)e * HP + colBase + brow) * DD) + bq;
  unsigned ldsB1[2], ldsB3[2];
#pragma unroll
  for (int b = 0; b < 2; ++b) {
    ldsB1[b] = lds_off32(&B1s[b][brow * BK + bq * 8]);
    ldsB3[b] = lds_off32(&B3s[b][brow * BK + bq * 8]);
  }
#endif

  // prologue: stage K-step 0 into buffer 0
  async_b128(ldsA[0],      asrc);
  async_b128(ldsA[0] + 16, asrc + 1);
#if HAVE_TDM
  if (tdmw) {
    tdm_load_2d(ldsB1buf[0], b1base, BN, BK, DD);
    tdm_load_2d(ldsB3buf[0], b3base, BN, BK, DD);
  }
#else
  async_b128(ldsB1[0], b1src);
  async_b128(ldsB3[0], b3src);
#endif

  f32x8 acc1[2][2] = {};
  f32x8 acc3[2][2] = {};

  constexpr int KT = DD / BK;
  for (int kt = 0; kt < KT; ++kt) {
    const int cur = kt & 1;
    if (kt + 1 < KT) {
      const int nxt = cur ^ 1;
      const int kofs = (kt + 1) * (BK / 8);   // uint4 units (8 bf16 each)
      async_b128(ldsA[nxt],      asrc + kofs);
      async_b128(ldsA[nxt] + 16, asrc + kofs + 1);
#if HAVE_TDM
      if (tdmw) {
        tdm_load_2d(ldsB1buf[nxt], b1base + (kt + 1) * BK, BN, BK, DD);
        tdm_load_2d(ldsB3buf[nxt], b3base + (kt + 1) * BK, BN, BK, DD);
      }
      wait_async<2>();              // drain CURRENT buffer's A copies (in-order)
      if (tdmw) wait_tensor<2>();   // drain CURRENT buffer's 2 TDM tiles
#else
      async_b128(ldsB1[nxt], b1src + kofs);
      async_b128(ldsB3[nxt], b3src + kofs);
      wait_async<4>();
#endif
    } else {
      wait_async<0>();
#if HAVE_TDM
      if (tdmw) wait_tensor<0>();
#endif
    }
    __syncthreads();

    // A fragment: 16-bit A 16x32 layout -> two contiguous 16B runs per lane
    bf16x16 af[2];
#pragma unroll
    for (int i = 0; i < 2; ++i) {
      const bf16x8* p = (const bf16x8*)(&As[cur][(wm + i * 16 + lrow) * BK + 8 * hi]);
      af[i] = cat8(p[0], p[2]);   // K = 8*hi..8*hi+7  and  16+8*hi..23+8*hi
    }
    // B fragment: K = 8*hi..8*hi+7 and 16+8*hi..23+8*hi (K-contiguous LDS rows)
    bf16x16 bf1[2], bf3[2];
#pragma unroll
    for (int j = 0; j < 2; ++j) {
      const int c = wn + j * 16 + lrow;
      const bf16x8* q1 = (const bf16x8*)(&B1s[cur][c * BK + 8 * hi]);
      const bf16x8* q3 = (const bf16x8*)(&B3s[cur][c * BK + 8 * hi]);
      bf1[j] = cat8(q1[0], q1[2]);
      bf3[j] = cat8(q3[0], q3[2]);
    }
#pragma unroll
    for (int i = 0; i < 2; ++i) {
#pragma unroll
      for (int j = 0; j < 2; ++j) {
        acc1[i][j] = __builtin_amdgcn_wmma_f32_16x16x32_bf16(
            false, af[i], false, bf1[j], (short)0, acc1[i][j], false, false);
        acc3[i][j] = __builtin_amdgcn_wmma_f32_16x16x32_bf16(
            false, af[i], false, bf3[j], (short)0, acc3[i][j], false, false);
      }
    }
    __syncthreads();   // protect buffer `cur` from next iteration's writes
  }

  // fused SwiGLU epilogue, store bf16 h rows (C/D layout: VGPR r -> M = r + 8*hi)
#pragma unroll
  for (int i = 0; i < 2; ++i) {
#pragma unroll
    for (int j = 0; j < 2; ++j) {
      const int col = colBase + wn + j * 16 + lrow;
#pragma unroll
      for (int r = 0; r < 8; ++r) {
        const int row = slotBase + wm + i * 16 + 8 * hi + r;
        const float g = acc1[i][j][r];
        const float u = acc3[i][j][r];
        const float hv = g * (1.f / (1.f + __expf(-g))) * u;
        hbuf[(size_t)row * HP + col] = (bf16_t)hv;
      }
    }
  }
}

// ---------------- pass 2: O_slot = h * W2  (grouped by expert) ----------------
__global__ __launch_bounds__(256) void k_ffn2(const bf16_t* __restrict__ hbuf,
                                              const bf16_t* __restrict__ W2bT,
                                              const int* __restrict__ tile_expert,
                                              float* __restrict__ Obuf) {
  __shared__ alignas(16) bf16_t As[2][BM * BK];
  __shared__ alignas(16) bf16_t Bs[2][BN * BK];

  const int tid = threadIdx.x;
  const int lane = tid & 31;
  const int wave = tid >> 5;
  const int wm = (wave & 3) * 32;
  const int wn = (wave >> 2) * 32;
  const int hi = lane >> 4;
  const int lrow = lane & 15;

  const int slotBase = blockIdx.x * BM;
  const int colBase = blockIdx.y * BN;
  const int e = tile_expert[blockIdx.x];

  const int arow = tid >> 1, ahalf = tid & 1;
  const uint4* __restrict__ asrc =
      (const uint4*)(hbuf + (size_t)(slotBase + arow) * HP) + ahalf * 2;

  unsigned ldsA[2];
#pragma unroll
  for (int b = 0; b < 2; ++b)
    ldsA[b] = lds_off32(&As[b][arow * BK + ahalf * 16]);

#if HAVE_TDM
  const bool tdmw = (wave == 0);
  const bf16_t* bbase = W2bT + ((size_t)e * DD + colBase) * HP;
  const unsigned ldsBbuf[2] = { lds_off32(&Bs[0][0]), lds_off32(&Bs[1][0]) };
#else
  const int brow = tid >> 2, bq = tid & 3;
  const uint4* __restrict__ bsrc =
      (const uint4*)(W2bT + ((size_t)e * DD + colBase + brow) * HP) + bq;
  unsigned ldsB[2];
#pragma unroll
  for (int b = 0; b < 2; ++b)
    ldsB[b] = lds_off32(&Bs[b][brow * BK + bq * 8]);
#endif

  async_b128(ldsA[0],      asrc);
  async_b128(ldsA[0] + 16, asrc + 1);
#if HAVE_TDM
  if (tdmw) tdm_load_2d(ldsBbuf[0], bbase, BN, BK, HP);
#else
  async_b128(ldsB[0], bsrc);
#endif

  f32x8 acc[2][2] = {};

  constexpr int KT = HP / BK;
  for (int kt = 0; kt < KT; ++kt) {
    const int cur = kt & 1;
    if (kt + 1 < KT) {
      const int nxt = cur ^ 1;
      const int kofs = (kt + 1) * (BK / 8);
      async_b128(ldsA[nxt],      asrc + kofs);
      async_b128(ldsA[nxt] + 16, asrc + kofs + 1);
#if HAVE_TDM
      if (tdmw) tdm_load_2d(ldsBbuf[nxt], bbase + (kt + 1) * BK, BN, BK, HP);
      wait_async<2>();
      if (tdmw) wait_tensor<1>();
#else
      async_b128(ldsB[nxt], bsrc + kofs);
      wait_async<3>();
#endif
    } else {
      wait_async<0>();
#if HAVE_TDM
      if (tdmw) wait_tensor<0>();
#endif
    }
    __syncthreads();

    bf16x16 af[2];
#pragma unroll
    for (int i = 0; i < 2; ++i) {
      const bf16x8* p = (const bf16x8*)(&As[cur][(wm + i * 16 + lrow) * BK + 8 * hi]);
      af[i] = cat8(p[0], p[2]);
    }
    bf16x16 bfj[2];
#pragma unroll
    for (int j = 0; j < 2; ++j) {
      const int c = wn + j * 16 + lrow;
      const bf16x8* q = (const bf16x8*)(&Bs[cur][c * BK + 8 * hi]);
      bfj[j] = cat8(q[0], q[2]);
    }
#pragma unroll
    for (int i = 0; i < 2; ++i) {
#pragma unroll
      for (int j = 0; j < 2; ++j) {
        acc[i][j] = __builtin_amdgcn_wmma_f32_16x16x32_bf16(
            false, af[i], false, bfj[j], (short)0, acc[i][j], false, false);
      }
    }
    __syncthreads();
  }

#pragma unroll
  for (int i = 0; i < 2; ++i) {
#pragma unroll
    for (int j = 0; j < 2; ++j) {
      const int col = colBase + wn + j * 16 + lrow;
#pragma unroll
      for (int r = 0; r < 8; ++r) {
        const int row = slotBase + wm + i * 16 + 8 * hi + r;
        Obuf[(size_t)row * DD + col] = acc[i][j][r];
      }
    }
  }
}

// ---------------- final gated combine (deterministic, no atomics) ----------------
__global__ __launch_bounds__(256) void k_combine(const float* __restrict__ Obuf,
                                                 const int* __restrict__ slot_idx,
                                                 const float* __restrict__ topv,
                                                 float* __restrict__ out) {
  const int i = blockIdx.x * 256 + threadIdx.x;   // over N*D
  const int n = i >> 10;
  const int d = i & (DD - 1);
  const int s0 = slot_idx[2 * n + 0];
  const int s1 = slot_idx[2 * n + 1];
  out[i] = topv[2 * n + 0] * Obuf[(size_t)s0 * DD + d] +
           topv[2 * n + 1] * Obuf[(size_t)s1 * DD + d];
}

// ---------------- host launcher ----------------
extern "C" void kernel_launch(void* const* d_in, const int* in_sizes, int n_in,
                              void* d_out, int out_size, void* d_ws, size_t ws_size,
                              hipStream_t stream) {
  (void)in_sizes; (void)n_in; (void)out_size; (void)ws_size;

  const float* x  = (const float*)d_in[0];   // [N, D]
  const float* Wr = (const float*)d_in[1];   // [D, E]
  const float* W1 = (const float*)d_in[2];   // [E, D, H]
  const float* W2 = (const float*)d_in[3];   // [E, H, D]
  const float* W3 = (const float*)d_in[4];   // [E, D, H]
  float* out = (float*)d_out;                // [N, D]

  char* ws = (char*)d_ws;
  size_t off = 0;
  auto carve = [&](size_t bytes) -> void* {
    void* p = ws + off;
    off = (off + bytes + 255) & ~(size_t)255;
    return p;
  };
  bf16_t* Xb        = (bf16_t*)carve((size_t)NN * DD * 2);
  bf16_t* W1bT      = (bf16_t*)carve((size_t)EE * HP * DD * 2);
  bf16_t* W3bT      = (bf16_t*)carve((size_t)EE * HP * DD * 2);
  bf16_t* W2bT      = (bf16_t*)carve((size_t)EE * DD * HP * 2);
  bf16_t* hbuf      = (bf16_t*)carve((size_t)CAP * HP * 2);
  float*  Obuf      = (float*)carve((size_t)CAP * DD * 4);
  int*    topi      = (int*)carve((size_t)NN * 2 * 4);
  float*  topv      = (float*)carve((size_t)NN * 2 * 4);
  int*    slot_tok  = (int*)carve((size_t)CAP * 4);
  int*    slot_idx  = (int*)carve((size_t)NN * 2 * 4);
  int*    tile_exp  = (int*)carve((size_t)NUM_MTILES * 4);

  k_convert_x<<<(NN * DD) / (256 * 4), 256, 0, stream>>>(x, Xb);
  k_convert_w13t<<<((long)EE * DD * HP) / 256, 256, 0, stream>>>(W1, W1bT);
  k_convert_w13t<<<((long)EE * DD * HP) / 256, 256, 0, stream>>>(W3, W3bT);
  k_convert_w2t<<<((long)EE * DD * HP) / 256, 256, 0, stream>>>(W2, W2bT);
  k_router<<<NN / 8, 256, 0, stream>>>(x, Wr, topi, topv);
  k_assign<<<1, 32, 0, stream>>>(topi, slot_tok, slot_idx, tile_exp);
  k_ffn1<<<dim3(NUM_MTILES, HP / BN), 256, 0, stream>>>(Xb, W1bT, W3bT, slot_tok,
                                                        tile_exp, hbuf);
  k_ffn2<<<dim3(NUM_MTILES, DD / BN), 256, 0, stream>>>(hbuf, W2bT, tile_exp, Obuf);
  k_combine<<<(NN * DD) / 256, 256, 0, stream>>>(Obuf, slot_idx, topv, out);
}
